// Seq2Seq_79173427134789
// MI455X (gfx1250) — matrix operand order repaired and checked by reading, not compiled, not run on hardware
//
#include <hip/hip_runtime.h>
#include <math.h>

// ---------------------------------------------------------------------------
// Seq2Seq (encoder RNN + attention decoder) for MI455X / gfx1250, wave32.
// All matmuls use v_wmma_f32_16x16x32_bf16 (bf16 inputs, f32 accumulate).
// GEMMs use 2x2 register blocking: 4 WMMAs per 8 b128 loads.
// ---------------------------------------------------------------------------

typedef __attribute__((ext_vector_type(16))) __bf16 v16bf;
typedef __attribute__((ext_vector_type(8)))  float  v8f;

#define B_   32
#define LS_  64
#define LT_  64
#define TD_  63          // decoder steps = LT-1
#define E_   512
#define H_   512
#define V_   32000

// ---- WMMA fragment helpers -------------------------------------------------
// A tile: 16x32 bf16, row-major source with leading dim `ld` (elements).
// Per ISA: lanes 0-15 hold row m, K = {k0..k0+7, k0+16..k0+23};
//          lanes 16-31 hold row m, K = {k0+8..k0+15, k0+24..k0+31}.
__device__ __forceinline__ v16bf load_a16(const __bf16* base, int ld,
                                          int m0, int k0, int lane) {
  int m  = m0 + (lane & 15);
  int kk = k0 + ((lane >> 4) << 3);
  const __bf16* p = base + (size_t)m * ld + kk;
  v16bf a;
  ((float4*)&a)[0] = *(const float4*)(p);        // k .. k+7
  ((float4*)&a)[1] = *(const float4*)(p + 16);   // k+16 .. k+23
  return a;
}

// B tile: 32x16 bf16 (KxN) from a transposed weight Wt[n][k] (leading dim ld).
// Per ISA: lanes 0-15 hold col n, K = {k0..k0+7, k0+16..k0+23};
//          lanes 16-31 hold col n, K = {k0+8..k0+15, k0+24..k0+31}.
__device__ __forceinline__ v16bf load_b16(const __bf16* wt, int ld,
                                          int n0, int k0, int lane) {
  int n  = n0 + (lane & 15);
  int kk = k0 + ((lane >> 4) << 3);
  const __bf16* p = wt + (size_t)n * ld + kk;
  v16bf b;
  ((float4*)&b)[0] = *(const float4*)(p);        // k .. k+7
  ((float4*)&b)[1] = *(const float4*)(p + 16);   // k+16 .. k+23
  return b;
}

__device__ __forceinline__ v8f wmma_bf16(v16bf a, v16bf b, v8f c) {
  return __builtin_amdgcn_wmma_f32_16x16x32_bf16(
      /*neg_a=*/false, a, /*neg_b=*/false, b,
      /*c_mod=*/(short)0, c, /*reuse_a=*/false, /*reuse_b=*/false);
}

// C/D 16x16 f32 tile store (+ optional biases): m = 8*(lane>=16)+i, n = lane&15
__device__ __forceinline__ void store_tile(float* C, int ldn, int m0, int n0,
                                           int lane, v8f acc,
                                           const float* bias0,
                                           const float* bias1) {
  int n  = n0 + (lane & 15);
  int mb = m0 + ((lane >> 4) << 3);
  float bs = (bias0 ? bias0[n] : 0.f) + (bias1 ? bias1[n] : 0.f);
  for (int i = 0; i < 8; i++) C[(size_t)(mb + i) * ldn + n] = acc[i] + bs;
}

// ---- Prep kernels ----------------------------------------------------------

// dst[n][k] = (bf16) src[k][n]   (src is KxN row-major fp32)
__global__ void k_transpose_cvt(const float* __restrict__ src,
                                __bf16* __restrict__ dst, int K, int N) {
  size_t total = (size_t)K * N;
  for (size_t i = (size_t)blockIdx.x * blockDim.x + threadIdx.x; i < total;
       i += (size_t)gridDim.x * blockDim.x) {
    size_t k = i / N, n = i % N;
    dst[n * (size_t)K + k] = (__bf16)src[i];
  }
}

// dst row (t*B + b) = (bf16) table[ids[b*Lids + t]]   (rows = Tused*B)
__global__ void k_embed_gather(const float* __restrict__ table,
                               const int* __restrict__ ids,
                               __bf16* __restrict__ dst, int rows, int Lids) {
  size_t total = (size_t)rows * E_;
  for (size_t i = (size_t)blockIdx.x * blockDim.x + threadIdx.x; i < total;
       i += (size_t)gridDim.x * blockDim.x) {
    int r = (int)(i / E_), e = (int)(i % E_);
    int t = r / B_, b = r % B_;
    int id = ids[b * Lids + t];
    dst[i] = (__bf16)table[(size_t)id * E_ + e];
  }
}

// ---- Generic WMMA GEMM: C[M][N] = A[M][K] @ Wt[N][K]^T + bias --------------
// block = 256 threads (8 waves). Wave w computes a 32x32 macro tile (2x2).
// grid = (N/256, M/32).
__global__ __launch_bounds__(256) void k_gemm_bf16(
    const __bf16* __restrict__ A, int K, const __bf16* __restrict__ Wt,
    const float* __restrict__ bias0, const float* __restrict__ bias1,
    float* __restrict__ C, int N) {
  int lane = threadIdx.x & 31, w = threadIdx.x >> 5;
  int m0 = blockIdx.y * 32;
  int n0 = (blockIdx.x * 16 + w * 2) * 16;
  v8f acc00 = {}, acc01 = {}, acc10 = {}, acc11 = {};
  for (int k0 = 0; k0 < K; k0 += 32) {
    v16bf a0 = load_a16(A, K, m0,      k0, lane);
    v16bf a1 = load_a16(A, K, m0 + 16, k0, lane);
    v16bf b0 = load_b16(Wt, K, n0,      k0, lane);
    v16bf b1 = load_b16(Wt, K, n0 + 16, k0, lane);
    acc00 = wmma_bf16(a0, b0, acc00);
    acc01 = wmma_bf16(a0, b1, acc01);
    acc10 = wmma_bf16(a1, b0, acc10);
    acc11 = wmma_bf16(a1, b1, acc11);
  }
  store_tile(C, N, m0,      n0,      lane, acc00, bias0, bias1);
  store_tile(C, N, m0,      n0 + 16, lane, acc01, bias0, bias1);
  store_tile(C, N, m0 + 16, n0,      lane, acc10, bias0, bias1);
  store_tile(C, N, m0 + 16, n0 + 16, lane, acc11, bias0, bias1);
}

// ---- Encoder: persistent single-WGP RNN scan -------------------------------
// h (32x512 bf16) lives in LDS; per step each of 32 waves computes both
// 16x16 m-tiles of its 16-column slab of h @ W_hh, sharing the B tile,
// adds the precomputed xproj (= x_t @ W_ih + b_ih + b_hh), applies tanh,
// rewrites LDS + globals.
__global__ __launch_bounds__(1024) void k_encoder(
    const __bf16* __restrict__ wt_hh, const float* __restrict__ xproj,
    float* __restrict__ hid_f, __bf16* __restrict__ hid_bf) {
  __shared__ __bf16 hbuf[B_ * H_];
  int tid = threadIdx.x, lane = tid & 31, w = tid >> 5;
  for (int i = tid; i < B_ * H_; i += 1024) hbuf[i] = (__bf16)0.f;
  __syncthreads();

  int n0 = w * 16;
  int n  = n0 + (lane & 15);
  int mb = (lane >> 4) << 3;          // 0 or 8
  for (int t = 0; t < LS_; t++) {
    v8f acc0 = {}, acc1 = {};
    for (int k0 = 0; k0 < H_; k0 += 32) {
      v16bf b  = load_b16(wt_hh, H_, n0, k0, lane);   // shared across m-tiles
      v16bf a0 = load_a16(hbuf, H_, 0,  k0, lane);
      v16bf a1 = load_a16(hbuf, H_, 16, k0, lane);
      acc0 = wmma_bf16(a0, b, acc0);
      acc1 = wmma_bf16(a1, b, acc1);
    }
    for (int i = 0; i < 8; i++) {
      acc0[i] = tanhf(acc0[i] + xproj[(size_t)(t * B_ + mb + i) * H_ + n]);
      acc1[i] = tanhf(acc1[i] + xproj[(size_t)(t * B_ + 16 + mb + i) * H_ + n]);
    }
    __syncthreads();                       // all reads of old h done
    for (int i = 0; i < 8; i++) {
      int m0 = mb + i, m1 = 16 + mb + i;
      float v0 = acc0[i], v1 = acc1[i];
      hbuf[m0 * H_ + n] = (__bf16)v0;
      hbuf[m1 * H_ + n] = (__bf16)v1;
      size_t r0 = (size_t)(t * B_ + m0) * H_ + n;
      size_t r1 = (size_t)(t * B_ + m1) * H_ + n;
      hid_f[r0] = v0;  hid_bf[r0] = (__bf16)v0;
      hid_f[r1] = v1;  hid_bf[r1] = (__bf16)v1;
    }
    __syncthreads();                       // new h visible
  }
}

// ---- Decoder recurrent step (attention + RNN cell), single workgroup -------
__global__ __launch_bounds__(1024) void k_dec_step(
    int t, const int* __restrict__ src_ids,
    const float* __restrict__ energy, const float* __restrict__ enc_hid_f,
    const float* __restrict__ h_in_f, const __bf16* __restrict__ h_in_bf,
    const __bf16* __restrict__ emb_tgt,
    const __bf16* __restrict__ wt_ih, const __bf16* __restrict__ wt_hh,
    const float* __restrict__ b_ih, const float* __restrict__ b_hh,
    float* __restrict__ h_out_f, __bf16* __restrict__ h_out_bf,
    __bf16* __restrict__ ctx_bf_out) {
  __shared__ float  sc[LS_ * B_];
  __shared__ float  wgt[LS_ * B_];
  __shared__ __bf16 ctx_lds[B_ * H_];
  int tid = threadIdx.x, lane = tid & 31, w = tid >> 5;

  // Phase A: attention scores[l][b] = energy[l,b,:] . h[b,:]  (masked)
  for (int p = tid; p < LS_ * B_; p += 1024) {
    int b = p & 31, l = p >> 5;
    const float* er = energy + (size_t)p * H_;
    const float* hr = h_in_f + (size_t)b * H_;
    float s = 0.f;
    for (int k = 0; k < H_; k++) s += er[k] * hr[k];
    sc[p] = (src_ids[b * LS_ + l] == 0) ? -1.0e9f : s;
  }
  __syncthreads();

  // Phase B: softmax over l for each b
  if (tid < B_) {
    int b = tid;
    float mx = -3.4e38f;
    for (int l = 0; l < LS_; l++) mx = fmaxf(mx, sc[l * B_ + b]);
    float sum = 0.f;
    for (int l = 0; l < LS_; l++) {
      float e = __expf(sc[l * B_ + b] - mx);
      wgt[l * B_ + b] = e; sum += e;
    }
    float inv = 1.f / sum;
    for (int l = 0; l < LS_; l++) wgt[l * B_ + b] *= inv;
  }
  __syncthreads();

  // Phase C: ctx[b][h] = sum_l w[b][l] * enc_hid[l,b,h]
  for (int i = tid; i < B_ * H_; i += 1024) {
    int b = i >> 9, h = i & (H_ - 1);
    float s = 0.f;
    for (int l = 0; l < LS_; l++)
      s += wgt[l * B_ + b] * enc_hid_f[(size_t)(l * B_ + b) * H_ + h];
    __bf16 cb = (__bf16)s;
    ctx_lds[i]    = cb;
    ctx_bf_out[i] = cb;     // for the logits GEMM kernel
  }
  __syncthreads();

  // Phase D: h_new = tanh([emb||ctx] @ W_ih + h @ W_hh + b_ih + b_hh)
  // Wave w owns a 16-column slab; B tiles shared across both m-tiles.
  const __bf16* emb = emb_tgt + (size_t)t * B_ * E_;
  int n0 = w * 16;
  v8f acc0 = {}, acc1 = {};
  for (int k0 = 0; k0 < E_ + H_; k0 += 32) {          // cat @ W_ih
    v16bf b = load_b16(wt_ih, E_ + H_, n0, k0, lane);
    v16bf a0, a1;
    if (k0 < E_) {
      a0 = load_a16(emb, E_, 0,  k0, lane);
      a1 = load_a16(emb, E_, 16, k0, lane);
    } else {
      a0 = load_a16(ctx_lds, H_, 0,  k0 - E_, lane);
      a1 = load_a16(ctx_lds, H_, 16, k0 - E_, lane);
    }
    acc0 = wmma_bf16(a0, b, acc0);
    acc1 = wmma_bf16(a1, b, acc1);
  }
  for (int k0 = 0; k0 < H_; k0 += 32) {               // + h @ W_hh
    v16bf b  = load_b16(wt_hh, H_, n0, k0, lane);
    v16bf a0 = load_a16(h_in_bf, H_, 0,  k0, lane);
    v16bf a1 = load_a16(h_in_bf, H_, 16, k0, lane);
    acc0 = wmma_bf16(a0, b, acc0);
    acc1 = wmma_bf16(a1, b, acc1);
  }
  int n = n0 + (lane & 15);
  float bs = b_ih[n] + b_hh[n];
  for (int i = 0; i < 8; i++) {
    acc0[i] = tanhf(acc0[i] + bs);
    acc1[i] = tanhf(acc1[i] + bs);
  }
  __syncthreads();      // all waves done reading h_in_bf (in-place update)
  int mb = (lane >> 4) << 3;
  for (int i = 0; i < 8; i++) {
    size_t o0 = (size_t)(mb + i) * H_ + n;
    size_t o1 = (size_t)(16 + mb + i) * H_ + n;
    h_out_f[o0]  = acc0[i];  h_out_bf[o0] = (__bf16)acc0[i];
    h_out_f[o1]  = acc1[i];  h_out_bf[o1] = (__bf16)acc1[i];
  }
}

// ---- Logits GEMM: out[b,t,:] = [h_new||ctx] @ h2o_w + h2o_b ----------------
// K = 1024, N = 32000, M = 32. Wave computes full M (2 m-tiles) x 2 n-tiles
// with 4 accumulators: 4 WMMAs per 8 b128 loads. Block = 8 waves = 256 cols.
// grid.x = V/256 = 125.
__global__ __launch_bounds__(256) void k_logits(
    int t, const __bf16* __restrict__ h_bf, const __bf16* __restrict__ ctx_bf,
    const __bf16* __restrict__ wt_h2o, const float* __restrict__ h2o_b,
    float* __restrict__ out) {
  int lane = threadIdx.x & 31, w = threadIdx.x >> 5;
  int n0 = (blockIdx.x * 16 + w * 2) * 16;
  v8f acc00 = {}, acc01 = {}, acc10 = {}, acc11 = {};
  for (int k0 = 0; k0 < 2 * H_; k0 += 32) {
    v16bf a0, a1;
    if (k0 < H_) {
      a0 = load_a16(h_bf, H_, 0,  k0, lane);
      a1 = load_a16(h_bf, H_, 16, k0, lane);
    } else {
      a0 = load_a16(ctx_bf, H_, 0,  k0 - H_, lane);
      a1 = load_a16(ctx_bf, H_, 16, k0 - H_, lane);
    }
    v16bf b0 = load_b16(wt_h2o, 2 * H_, n0,      k0, lane);
    v16bf b1 = load_b16(wt_h2o, 2 * H_, n0 + 16, k0, lane);
    acc00 = wmma_bf16(a0, b0, acc00);
    acc01 = wmma_bf16(a0, b1, acc01);
    acc10 = wmma_bf16(a1, b0, acc10);
    acc11 = wmma_bf16(a1, b1, acc11);
  }
  int mb = (lane >> 4) << 3;
  for (int jt = 0; jt < 2; jt++) {
    int n = n0 + jt * 16 + (lane & 15);
    float bs = h2o_b[n];
    v8f lo = jt ? acc01 : acc00;
    v8f hi = jt ? acc11 : acc10;
    for (int i = 0; i < 8; i++) {
      int b0r = mb + i, b1r = 16 + mb + i;
      out[((size_t)b0r * TD_ + t) * V_ + n] = lo[i] + bs;
      out[((size_t)b1r * TD_ + t) * V_ + n] = hi[i] + bs;
    }
  }
}

// ---- Fused log-softmax over V per (b,t) row, in place ----------------------
__global__ __launch_bounds__(256) void k_logsoftmax(int t, float* __restrict__ out) {
  __shared__ float red[256];
  int b = blockIdx.x, tid = threadIdx.x;
  float* row = out + ((size_t)b * TD_ + t) * V_;
  float mx = -3.4e38f;
  for (int i = tid; i < V_; i += 256) mx = fmaxf(mx, row[i]);
  red[tid] = mx; __syncthreads();
  for (int s = 128; s > 0; s >>= 1) {
    if (tid < s) red[tid] = fmaxf(red[tid], red[tid + s]);
    __syncthreads();
  }
  mx = red[0]; __syncthreads();
  float sum = 0.f;
  for (int i = tid; i < V_; i += 256) sum += __expf(row[i] - mx);
  red[tid] = sum; __syncthreads();
  for (int s = 128; s > 0; s >>= 1) {
    if (tid < s) red[tid] += red[tid + s];
    __syncthreads();
  }
  float lse = mx + __logf(red[0]);
  __syncthreads();
  for (int i = tid; i < V_; i += 256) row[i] -= lse;
}

// ---------------------------------------------------------------------------
extern "C" void kernel_launch(void* const* d_in, const int* in_sizes, int n_in,
                              void* d_out, int out_size, void* d_ws, size_t ws_size,
                              hipStream_t stream) {
  (void)in_sizes; (void)n_in; (void)out_size; (void)ws_size;
  const int*   src       = (const int*)  d_in[0];
  const int*   tgt       = (const int*)  d_in[1];
  const float* enc_embed = (const float*)d_in[2];
  const float* enc_w_ih  = (const float*)d_in[3];
  const float* enc_b_ih  = (const float*)d_in[4];
  const float* enc_w_hh  = (const float*)d_in[5];
  const float* enc_b_hh  = (const float*)d_in[6];
  const float* dec_embed = (const float*)d_in[7];
  const float* attn_w    = (const float*)d_in[8];
  const float* attn_b    = (const float*)d_in[9];
  const float* dec_w_ih  = (const float*)d_in[10];
  const float* dec_b_ih  = (const float*)d_in[11];
  const float* dec_w_hh  = (const float*)d_in[12];
  const float* dec_b_hh  = (const float*)d_in[13];
  const float* h2o_w     = (const float*)d_in[14];
  const float* h2o_b     = (const float*)d_in[15];
  float* out = (float*)d_out;

  // Workspace carve-out (~88 MB, 256B aligned)
  char* p = (char*)d_ws;
  auto alloc = [&](size_t bytes) -> char* {
    char* r = p; p += (bytes + 255) & ~(size_t)255; return r;
  };
  __bf16* wt_enc_ih  = (__bf16*)alloc((size_t)H_ * E_ * 2);
  __bf16* wt_enc_hh  = (__bf16*)alloc((size_t)H_ * H_ * 2);
  __bf16* wt_attn    = (__bf16*)alloc((size_t)H_ * H_ * 2);
  __bf16* wt_dec_ih  = (__bf16*)alloc((size_t)H_ * (E_ + H_) * 2);
  __bf16* wt_dec_hh  = (__bf16*)alloc((size_t)H_ * H_ * 2);
  __bf16* wt_h2o     = (__bf16*)alloc((size_t)V_ * (2 * H_) * 2);   // 62.5 MB
  __bf16* a_src      = (__bf16*)alloc((size_t)B_ * LS_ * E_ * 2);
  float*  xproj      = (float*) alloc((size_t)B_ * LS_ * H_ * 4);
  float*  enc_hid_f  = (float*) alloc((size_t)B_ * LS_ * H_ * 4);
  __bf16* enc_hid_bf = (__bf16*)alloc((size_t)B_ * LS_ * H_ * 2);
  float*  energy_f   = (float*) alloc((size_t)B_ * LS_ * H_ * 4);
  __bf16* emb_tgt    = (__bf16*)alloc((size_t)B_ * TD_ * E_ * 2);
  float*  h_f        = (float*) alloc((size_t)B_ * H_ * 4);
  __bf16* h_bf       = (__bf16*)alloc((size_t)B_ * H_ * 2);
  __bf16* ctx_bf     = (__bf16*)alloc((size_t)B_ * H_ * 2);

  dim3 tb(256);
  // One-time: bf16 weight transposes (B-operands become contiguous b128 loads)
  k_transpose_cvt<<<dim3(512),  tb, 0, stream>>>(enc_w_ih, wt_enc_ih, E_, H_);
  k_transpose_cvt<<<dim3(512),  tb, 0, stream>>>(enc_w_hh, wt_enc_hh, H_, H_);
  k_transpose_cvt<<<dim3(512),  tb, 0, stream>>>(attn_w,   wt_attn,   H_, H_);
  k_transpose_cvt<<<dim3(1024), tb, 0, stream>>>(dec_w_ih, wt_dec_ih, E_ + H_, H_);
  k_transpose_cvt<<<dim3(512),  tb, 0, stream>>>(dec_w_hh, wt_dec_hh, H_, H_);
  k_transpose_cvt<<<dim3(8192), tb, 0, stream>>>(h2o_w,    wt_h2o,    2 * H_, V_);
  // Embedding gathers (fp32 table -> bf16 rows, time-major)
  k_embed_gather<<<dim3(2048), tb, 0, stream>>>(enc_embed, src, a_src,  B_ * LS_, LS_);
  k_embed_gather<<<dim3(2048), tb, 0, stream>>>(dec_embed, tgt, emb_tgt, B_ * TD_, LT_);
  // Hoisted input projection: xproj = src_emb @ W_ih + b_ih + b_hh (2048x512x512)
  k_gemm_bf16<<<dim3(H_ / 256, (B_ * LS_) / 32), tb, 0, stream>>>(
      a_src, E_, wt_enc_ih, enc_b_ih, enc_b_hh, xproj, H_);
  // Encoder scan (persistent single workgroup, 32 waves)
  k_encoder<<<dim3(1), dim3(1024), 0, stream>>>(wt_enc_hh, xproj, enc_hid_f, enc_hid_bf);
  // Hoisted attention projection: energy = enc_hid @ attn_w + attn_b
  k_gemm_bf16<<<dim3(H_ / 256, (B_ * LS_) / 32), tb, 0, stream>>>(
      enc_hid_bf, H_, wt_attn, attn_b, nullptr, energy_f, H_);

  // Decoder: 63 sequential steps (attention+cell -> logits GEMM -> log-softmax)
  for (int t = 0; t < TD_; t++) {
    const float*  hin_f = (t == 0) ? enc_hid_f  + (size_t)(LS_ - 1) * B_ * H_ : h_f;
    const __bf16* hin_b = (t == 0) ? enc_hid_bf + (size_t)(LS_ - 1) * B_ * H_ : h_bf;
    k_dec_step<<<dim3(1), dim3(1024), 0, stream>>>(
        t, src, energy_f, enc_hid_f, hin_f, hin_b, emb_tgt,
        wt_dec_ih, wt_dec_hh, dec_b_ih, dec_b_hh, h_f, h_bf, ctx_bf);
    k_logits<<<dim3(V_ / 256), tb, 0, stream>>>(t, h_bf, ctx_bf, wt_h2o, h2o_b, out);
    k_logsoftmax<<<dim3(B_), tb, 0, stream>>>(t, out);
  }
}
